// Rs_GCN_76072460747469
// MI455X (gfx1250) — compile-verified, matrix operand order checked by reading, not run on hardware
//
#include <hip/hip_runtime.h>
#include <hip/hip_bf16.h>
#include <stdint.h>

// ---------------- problem constants ----------------
#define kB   8
#define kD   1024
#define kN   1024
#define kI   512      // INTER
#define kH   16
#define kHD  32
#define kEPS 1e-5f

typedef __bf16 bf16_t;
typedef __attribute__((ext_vector_type(16))) __bf16 bf16x16;
typedef __attribute__((ext_vector_type(8)))  float  floatx8;

union Frag {
  bf16x16 v;
  uint4 q[2];
};
union Pack8 {
  bf16_t h[8];
  uint4 q;
};
union Pack4 {
  bf16_t h[4];
  uint2 q;
};

__device__ __forceinline__ floatx8 fzero() {
  floatx8 z = {0.f, 0.f, 0.f, 0.f, 0.f, 0.f, 0.f, 0.f};
  return z;
}

// CDNA5 async copy: 16 bytes global -> LDS per lane, tracked by ASYNCcnt.
// LDS destination offset comes from a VGPR (wave-relative, low 32 bits of
// the flat LDS pointer per ISA 10.2).
__device__ __forceinline__ uint32_t lds_off(const void* p) {
  return (uint32_t)(size_t)p;
}
__device__ __forceinline__ void async_copy_b128(uint32_t lds_byte, const void* g) {
  asm volatile("global_load_async_to_lds_b128 %0, %1, off"
               :: "v"(lds_byte), "v"((uint64_t)(size_t)g)
               : "memory");
}
__device__ __forceinline__ void wait_async_all() {
  asm volatile("s_wait_asynccnt 0x0" ::: "memory");
}
// Async loads retire in order: cnt<=6 means the previous 6-op group landed
// while the newest 6-op group keeps streaming (double-buffer overlap).
__device__ __forceinline__ void wait_async_prev6() {
  asm volatile("s_wait_asynccnt 0x6" ::: "memory");
}

// 16x32 bf16 WMMA operand fragment from a row-major [rows x 32] tile.
// ISA layout: lanes 0-15 -> row (base+lane), K = {0..7, 16..23}
//             lanes 16-31 -> row (base+lane-16), K = {8..15, 24..31}
__device__ __forceinline__ Frag load_frag(const bf16_t* __restrict__ base,
                                          int stride, int rowBase, int lane) {
  Frag f;
  const int r   = rowBase + (lane & 15);
  const int off = (lane < 16) ? 0 : 8;
  const bf16_t* p = base + (size_t)r * stride + off;
  f.q[0] = *reinterpret_cast<const uint4*>(p);
  f.q[1] = *reinterpret_cast<const uint4*>(p + 16);
  return f;
}

__device__ __forceinline__ floatx8 wmma_bf16(const Frag& a, const Frag& b, floatx8 c) {
  return __builtin_amdgcn_wmma_f32_16x16x32_bf16(false, a.v, false, b.v,
                                                 (short)0, c, false, false);
}

// ---------------- kernel: elementwise f32 -> bf16 --------------------------
__global__ __launch_bounds__(256)
void cvt_kernel(const float* __restrict__ src, bf16_t* __restrict__ dst, int n) {
  const int i = (blockIdx.x * 256 + threadIdx.x) * 4;
  if (i + 3 < n) {
    const float4 f = *reinterpret_cast<const float4*>(src + i);
    Pack4 u;
    u.h[0] = (bf16_t)f.x; u.h[1] = (bf16_t)f.y;
    u.h[2] = (bf16_t)f.z; u.h[3] = (bf16_t)f.w;
    *reinterpret_cast<uint2*>(dst + i) = u.q;
  }
}

// ---------------- kernel: vT[b][n][d] = (bf16) v[b][d][n] ------------------
__global__ __launch_bounds__(256)
void transpose_cvt_kernel(const float* __restrict__ v, bf16_t* __restrict__ vT) {
  __shared__ float tile[64][65];   // +1 pad: no bank conflicts
  const int b  = blockIdx.z;
  const int d0 = blockIdx.y * 64;
  const int n0 = blockIdx.x * 64;
  const int t  = threadIdx.x;
  const int r  = t >> 2;           // 0..63
  const int c0 = (t & 3) * 16;
#pragma unroll
  for (int e = 0; e < 16; ++e)
    tile[r][c0 + e] = v[((size_t)b * kD + d0 + r) * kN + n0 + c0 + e];
  __syncthreads();
#pragma unroll
  for (int e = 0; e < 16; ++e)
    vT[((size_t)b * kN + n0 + r) * kD + d0 + c0 + e] = (bf16_t)tile[c0 + e][r];
}

// ---------------- kernel: rmask[b][n] = mask[b][n] / sum_n mask ------------
__global__ __launch_bounds__(256)
void rmask_kernel(const float* __restrict__ mask, float* __restrict__ rmask) {
  __shared__ float red[256];
  const int b = blockIdx.x, t = threadIdx.x;
  float s = 0.f;
  for (int i = t; i < kN; i += 256) s += mask[b * kN + i];
  red[t] = s;
  __syncthreads();
  for (int o = 128; o > 0; o >>= 1) {
    if (t < o) red[t] += red[t + o];
    __syncthreads();
  }
  const float invC = 1.0f / red[0];
  for (int i = t; i < kN; i += 256) rmask[b * kN + i] = mask[b * kN + i] * invC;
}

// ---------------- kernel: projection GEMM  out = W @ v_b + bias ------------
// A = Wbf (kI x kD bf16, row-major), B = vT (kN x kD bf16, node-major).
// Block tile 128(M) x 64(N), 4 waves, each a 32x64 strip: 8 WMMA / K-step.
// Double-buffered LDS; async group k+1 streams while WMMAs consume group k.
// mode 0: store (B,INTER,N) bf16            (phi)
// mode 1: store (B,INTER,N) bf16 * mask[n]  (g)
// mode 2: store (B,N,INTER) bf16 transposed (theta -> contiguous b128 stores)
__global__ __launch_bounds__(128)
void proj_kernel(const bf16_t* __restrict__ vT, const bf16_t* __restrict__ Wbf,
                 const float* __restrict__ bias, const float* __restrict__ mask,
                 bf16_t* __restrict__ out, int mode) {
  __shared__ bf16_t As[2][128 * 32];   // [m][k] ping-pong
  __shared__ bf16_t Bs[2][64 * 32];    // [n][k] ping-pong
  __shared__ float  biasS[128];

  const int b     = blockIdx.z;
  const int mBase = blockIdx.y * 128;
  const int nBase = blockIdx.x * 64;
  const int t     = threadIdx.x;
  const int lane  = t & 31;
  const int wid   = t >> 5;

  biasS[t] = bias[mBase + t];

  const uint32_t ldsA = lds_off(As);
  const uint32_t ldsB = lds_off(Bs);

  // per-thread staging coordinates (row r, element chunk c)
  const int rA = t >> 2, cA = (t & 3) * 8;           // A: 4 groups of 32 rows
  const int rB = t >> 2, cB = (t & 3) * 8;           // B: 2 groups of 32 rows
  const bf16_t* gA = Wbf + (size_t)(mBase + rA) * kD + cA;
  const bf16_t* gB = vT + ((size_t)b * kN + nBase + rB) * kD + cB;

  // issue one 6-op async group (4x A rows, 2x B rows) for K-slab k0
  auto issue = [&](int buf, int k0) {
    const uint32_t a0 = ldsA + (uint32_t)buf * (128 * 32 * 2);
    const uint32_t b0 = ldsB + (uint32_t)buf * (64 * 32 * 2);
#pragma unroll
    for (int i = 0; i < 4; ++i)
      async_copy_b128(a0 + (uint32_t)((rA + 32 * i) * 32 + cA) * 2,
                      gA + (size_t)(32 * i) * kD + k0);
#pragma unroll
    for (int i = 0; i < 2; ++i)
      async_copy_b128(b0 + (uint32_t)((rB + 32 * i) * 32 + cB) * 2,
                      gB + (size_t)(32 * i) * kD + k0);
  };

  floatx8 acc[2][4];
#pragma unroll
  for (int i = 0; i < 2; ++i)
#pragma unroll
    for (int j = 0; j < 4; ++j) acc[i][j] = fzero();

  const int mT = wid * 32;
  const int kSteps = kD / 32;

  issue(0, 0);
  for (int it = 0; it < kSteps; ++it) {
    const int cur = it & 1;
    if (it + 1 < kSteps) {
      issue(cur ^ 1, (it + 1) * 32);
      wait_async_prev6();      // group `it` landed; group it+1 keeps streaming
    } else {
      wait_async_all();
    }
    __syncthreads();

    const bf16_t* Ac = As[cur];
    const bf16_t* Bc = Bs[cur];
    Frag a0 = load_frag(Ac, 32, mT, lane);
    Frag a1 = load_frag(Ac, 32, mT + 16, lane);
#pragma unroll
    for (int j = 0; j < 4; ++j) {
      Frag bj = load_frag(Bc, 32, j * 16, lane);
      acc[0][j] = wmma_bf16(a0, bj, acc[0][j]);
      acc[1][j] = wmma_bf16(a1, bj, acc[1][j]);
    }
    __syncthreads();   // all readers done before this buffer is refilled
  }

  // Epilogue. C/D layout: lane 0-15: N=lane, M=i; lane 16-31: N=lane-16, M=i+8
  const int colL = lane & 15;
  const int hi8  = (lane >> 4) * 8;
#pragma unroll
  for (int ti = 0; ti < 2; ++ti) {
#pragma unroll
    for (int tj = 0; tj < 4; ++tj) {
      const int m0 = mBase + mT + ti * 16 + hi8;
      const int n  = nBase + tj * 16 + colL;
      const floatx8 a = acc[ti][tj];
      if (mode == 2) {
        Pack8 u;
#pragma unroll
        for (int e = 0; e < 8; ++e)
          u.h[e] = (bf16_t)(a[e] + biasS[m0 - mBase + e]);
        *reinterpret_cast<uint4*>(out + ((size_t)b * kN + n) * kI + m0) = u.q;
      } else {
        const float mk = (mode == 1) ? mask[b * kN + n] : 1.0f;
#pragma unroll
        for (int e = 0; e < 8; ++e)
          out[((size_t)b * kI + m0 + e) * kN + n] =
              (bf16_t)((a[e] + biasS[m0 - mBase + e]) * mk);
      }
    }
  }
}

// ---------------- kernel: per-(b,h) Gram matrix S = phi_h^T @ g_h ----------
// phi,g (B,INTER,N) bf16; one wave per (b,h); output S^T (B,H,32,32) bf16.
__global__ __launch_bounds__(32)
void gram_kernel(const bf16_t* __restrict__ phi, const bf16_t* __restrict__ g,
                 bf16_t* __restrict__ ST) {
  __shared__ bf16_t Ps[32 * 32];
  __shared__ bf16_t Gs[32 * 32];
  const int bh = blockIdx.x;
  const int b = bh / kH, h = bh % kH;
  const int lane = threadIdx.x;

  const bf16_t* prow = phi + ((size_t)b * kI + h * kHD + lane) * kN;
  const bf16_t* grow = g   + ((size_t)b * kI + h * kHD + lane) * kN;
  const uint32_t ldsP = lds_off(Ps) + (uint32_t)lane * 64;
  const uint32_t ldsG = lds_off(Gs) + (uint32_t)lane * 64;

  floatx8 acc[2][2];
  acc[0][0] = fzero(); acc[0][1] = fzero();
  acc[1][0] = fzero(); acc[1][1] = fzero();

  for (int k0 = 0; k0 < kN; k0 += 32) {
#pragma unroll
    for (int c = 0; c < 4; ++c) {
      async_copy_b128(ldsP + c * 16, prow + k0 + c * 8);
      async_copy_b128(ldsG + c * 16, grow + k0 + c * 8);
    }
    wait_async_all();
    __syncthreads();
    Frag p0 = load_frag(Ps, 32, 0, lane);
    Frag p1 = load_frag(Ps, 32, 16, lane);
    Frag g0 = load_frag(Gs, 32, 0, lane);
    Frag g1 = load_frag(Gs, 32, 16, lane);
    acc[0][0] = wmma_bf16(p0, g0, acc[0][0]);
    acc[0][1] = wmma_bf16(p0, g1, acc[0][1]);
    acc[1][0] = wmma_bf16(p1, g0, acc[1][0]);
    acc[1][1] = wmma_bf16(p1, g1, acc[1][1]);
    __syncthreads();
  }

  const int colL = lane & 15;
  const int hi8  = (lane >> 4) * 8;
  bf16_t* stb = ST + (size_t)bh * (kHD * kHD);
#pragma unroll
  for (int ti = 0; ti < 2; ++ti) {
#pragma unroll
    for (int tj = 0; tj < 2; ++tj) {
      const int i0 = ti * 16 + hi8;       // M index (phi dim i)
      const int j  = tj * 16 + colL;      // N index (g dim j)
      Pack8 u;
#pragma unroll
      for (int e = 0; e < 8; ++e) u.h[e] = (bf16_t)acc[ti][tj][e];
      *reinterpret_cast<uint4*>(stb + j * kHD + i0) = u.q;  // S^T[j][i]
    }
  }
}

// ---------------- kernel: y = (mask/C) * theta_h @ S_h ---------------------
// thetaT (B,N,INTER) bf16, ST (B,H,32,32) bf16 transposed.
// Output y node-major (B,N,INTER) so the final GEMM can async-stage it.
__global__ __launch_bounds__(128)
void apply_kernel(const bf16_t* __restrict__ thetaT, const bf16_t* __restrict__ ST,
                  const float* __restrict__ rmask, bf16_t* __restrict__ y) {
  const int bh = blockIdx.y;
  const int b = bh / kH, h = bh % kH;
  const int t = threadIdx.x;
  const int lane = t & 31, wid = t >> 5;
  const int nBase = blockIdx.x * 128 + wid * 32;

  const bf16_t* stb = ST + (size_t)bh * (kHD * kHD);
  Frag s0 = load_frag(stb, kHD, 0, lane);   // columns of S = rows of S^T
  Frag s1 = load_frag(stb, kHD, 16, lane);

  const bf16_t* thb = thetaT + (size_t)b * kN * kI + h * kHD;
  Frag a0 = load_frag(thb, kI, nBase, lane);
  Frag a1 = load_frag(thb, kI, nBase + 16, lane);

  floatx8 acc[2][2];
  acc[0][0] = wmma_bf16(a0, s0, fzero());
  acc[0][1] = wmma_bf16(a0, s1, fzero());
  acc[1][0] = wmma_bf16(a1, s0, fzero());
  acc[1][1] = wmma_bf16(a1, s1, fzero());

  const int colL = lane & 15;
  const int hi8  = (lane >> 4) * 8;
#pragma unroll
  for (int ti = 0; ti < 2; ++ti) {
#pragma unroll
    for (int tj = 0; tj < 2; ++tj) {
      const int j  = h * kHD + tj * 16 + colL;   // channel index
      const int n0 = nBase + ti * 16 + hi8;      // node index (varies over e)
      const float* rm = rmask + b * kN + n0;
#pragma unroll
      for (int e = 0; e < 8; ++e)
        y[((size_t)b * kN + n0 + e) * kI + j] = (bf16_t)(acc[ti][tj][e] * rm[e]);
    }
  }
}

// ---------------- kernel: out = BN(Ww @ y + bw) * mask + v -----------------
// A = Wwbf (kD x kI bf16), B = y (kN x kI bf16 node-major); double-buffered
// async staging identical to proj_kernel.
__global__ __launch_bounds__(128)
void out_kernel(const bf16_t* __restrict__ Wwbf, const bf16_t* __restrict__ y,
                const float* __restrict__ bw, const float* __restrict__ gamma,
                const float* __restrict__ beta, const float* __restrict__ mean,
                const float* __restrict__ var, const float* __restrict__ mask,
                const float* __restrict__ v, float* __restrict__ out) {
  __shared__ bf16_t As[2][128 * 32];
  __shared__ bf16_t Bs[2][64 * 32];
  __shared__ float scaleS[128], shiftS[128];

  const int b     = blockIdx.z;
  const int mBase = blockIdx.y * 128;
  const int nBase = blockIdx.x * 64;
  const int t     = threadIdx.x;
  const int lane  = t & 31;
  const int wid   = t >> 5;

  {
    const int m = mBase + t;
    const float inv = gamma[m] * rsqrtf(var[m] + kEPS);
    scaleS[t] = inv;
    shiftS[t] = (bw[m] - mean[m]) * inv + beta[m];
  }

  const uint32_t ldsA = lds_off(As);
  const uint32_t ldsB = lds_off(Bs);

  const int rA = t >> 2, cA = (t & 3) * 8;
  const int rB = t >> 2, cB = (t & 3) * 8;
  const bf16_t* gA = Wwbf + (size_t)(mBase + rA) * kI + cA;
  const bf16_t* gB = y + ((size_t)b * kN + nBase + rB) * kI + cB;

  auto issue = [&](int buf, int k0) {
    const uint32_t a0 = ldsA + (uint32_t)buf * (128 * 32 * 2);
    const uint32_t b0 = ldsB + (uint32_t)buf * (64 * 32 * 2);
#pragma unroll
    for (int i = 0; i < 4; ++i)
      async_copy_b128(a0 + (uint32_t)((rA + 32 * i) * 32 + cA) * 2,
                      gA + (size_t)(32 * i) * kI + k0);
#pragma unroll
    for (int i = 0; i < 2; ++i)
      async_copy_b128(b0 + (uint32_t)((rB + 32 * i) * 32 + cB) * 2,
                      gB + (size_t)(32 * i) * kI + k0);
  };

  floatx8 acc[2][4];
#pragma unroll
  for (int i = 0; i < 2; ++i)
#pragma unroll
    for (int j = 0; j < 4; ++j) acc[i][j] = fzero();

  const int mT = wid * 32;
  const int kSteps = kI / 32;

  issue(0, 0);
  for (int it = 0; it < kSteps; ++it) {
    const int cur = it & 1;
    if (it + 1 < kSteps) {
      issue(cur ^ 1, (it + 1) * 32);
      wait_async_prev6();
    } else {
      wait_async_all();
    }
    __syncthreads();

    const bf16_t* Ac = As[cur];
    const bf16_t* Bc = Bs[cur];
    Frag a0 = load_frag(Ac, 32, mT, lane);
    Frag a1 = load_frag(Ac, 32, mT + 16, lane);
#pragma unroll
    for (int j = 0; j < 4; ++j) {
      Frag bj = load_frag(Bc, 32, j * 16, lane);
      acc[0][j] = wmma_bf16(a0, bj, acc[0][j]);
      acc[1][j] = wmma_bf16(a1, bj, acc[1][j]);
    }
    __syncthreads();
  }

  const int colL = lane & 15;
  const int hi8  = (lane >> 4) * 8;
#pragma unroll
  for (int ti = 0; ti < 2; ++ti) {
#pragma unroll
    for (int tj = 0; tj < 4; ++tj) {
      const int m0 = mBase + mT + ti * 16 + hi8;
      const int n  = nBase + tj * 16 + colL;
      const float mk = mask[b * kN + n];
#pragma unroll
      for (int e = 0; e < 8; ++e) {
        const int m = m0 + e;
        float val = acc[ti][tj][e] * scaleS[m - mBase] + shiftS[m - mBase];
        val = val * mk + v[((size_t)b * kD + m) * kN + n];
        out[((size_t)b * kD + m) * kN + n] = val;
      }
    }
  }
}

// ---------------- host launch ----------------------------------------------
extern "C" void kernel_launch(void* const* d_in, const int* in_sizes, int n_in,
                              void* d_out, int out_size, void* d_ws, size_t ws_size,
                              hipStream_t stream) {
  const float* v    = (const float*)d_in[0];
  const float* mask = (const float*)d_in[1];
  const float* Wg   = (const float*)d_in[2];
  const float* bg   = (const float*)d_in[3];
  const float* Wt   = (const float*)d_in[4];
  const float* bt   = (const float*)d_in[5];
  const float* Wp   = (const float*)d_in[6];
  const float* bp   = (const float*)d_in[7];
  const float* Ww   = (const float*)d_in[8];
  const float* bw   = (const float*)d_in[9];
  const float* gma  = (const float*)d_in[10];
  const float* bta  = (const float*)d_in[11];
  const float* mu   = (const float*)d_in[12];
  const float* var  = (const float*)d_in[13];
  float* out = (float*)d_out;

  char* ws = (char*)d_ws;
  size_t off = 0;
  auto carve = [&](size_t bytes) -> void* {
    void* p = ws + off;
    off = (off + bytes + 255) & ~(size_t)255;
    return p;
  };
  bf16_t* vT     = (bf16_t*)carve((size_t)kB * kN * kD * sizeof(bf16_t));
  bf16_t* Wgbf   = (bf16_t*)carve((size_t)kI * kD * sizeof(bf16_t));
  bf16_t* Wtbf   = (bf16_t*)carve((size_t)kI * kD * sizeof(bf16_t));
  bf16_t* Wpbf   = (bf16_t*)carve((size_t)kI * kD * sizeof(bf16_t));
  bf16_t* Wwbf   = (bf16_t*)carve((size_t)kD * kI * sizeof(bf16_t));
  bf16_t* thetaT = (bf16_t*)carve((size_t)kB * kN * kI * sizeof(bf16_t));
  bf16_t* phi    = (bf16_t*)carve((size_t)kB * kI * kN * sizeof(bf16_t));
  bf16_t* g      = (bf16_t*)carve((size_t)kB * kI * kN * sizeof(bf16_t));
  bf16_t* y      = (bf16_t*)carve((size_t)kB * kN * kI * sizeof(bf16_t));
  bf16_t* ST     = (bf16_t*)carve((size_t)kB * kH * kHD * kHD * sizeof(bf16_t));
  float*  rmask  = (float*)carve((size_t)kB * kN * sizeof(float));
  (void)ws_size; (void)in_sizes; (void)n_in; (void)out_size;

  // one-off conversions (all row-copy staging afterwards)
  const int wElems = kI * kD;
  cvt_kernel<<<wElems / 1024, 256, 0, stream>>>(Wg, Wgbf, wElems);
  cvt_kernel<<<wElems / 1024, 256, 0, stream>>>(Wt, Wtbf, wElems);
  cvt_kernel<<<wElems / 1024, 256, 0, stream>>>(Wp, Wpbf, wElems);
  cvt_kernel<<<wElems / 1024, 256, 0, stream>>>(Ww, Wwbf, wElems);
  transpose_cvt_kernel<<<dim3(kN / 64, kD / 64, kB), 256, 0, stream>>>(v, vT);
  rmask_kernel<<<kB, 256, 0, stream>>>(mask, rmask);

  const dim3 gp(kN / 64, kI / 128, kB);
  proj_kernel<<<gp, 128, 0, stream>>>(vT, Wtbf, bt, mask, thetaT, 2);  // theta^T
  proj_kernel<<<gp, 128, 0, stream>>>(vT, Wpbf, bp, mask, phi, 0);     // phi
  proj_kernel<<<gp, 128, 0, stream>>>(vT, Wgbf, bg, mask, g, 1);       // g*mask

  gram_kernel<<<kB * kH, 32, 0, stream>>>(phi, g, ST);

  apply_kernel<<<dim3(kN / 128, kB * kH), 128, 0, stream>>>(thetaT, ST, rmask, y);

  out_kernel<<<dim3(kN / 64, kD / 128, kB), 128, 0, stream>>>(
      Wwbf, y, bw, gma, bta, mu, var, mask, v, out);
}
